// GLAWorkingMemory_14594298872483
// MI455X (gfx1250) — compile-verified, matrix operand order checked by reading, not run on hardware
//
#include <hip/hip_runtime.h>
#include <hip/hip_bf16.h>

// ---------------------------------------------------------------------------
// GLA working-memory layer for MI455X (gfx1250, wave32, WMMA bf16 path)
//   q,k,v = x@W^T+b ; g = logsigmoid((x@Wg1^T)@Wg2^T + bg2)/16
//   scan:  S = S*keep*exp(g) + k (outer) v ; out = q . S
//   y = out@Wo^T + bo ; outputs (y, new_state) concatenated in d_out.
//
// GEMMs: double-buffered LDS staging via GLOBAL_LOAD_ASYNC_TO_LDS_B128
// (ASYNCcnt-tracked, overlapped with v_wmma_f32_16x16x32_bf16 compute).
// ---------------------------------------------------------------------------

#define BS   32
#define PP   512
#define DD   2048
#define DWM  2048
#define HH   16
#define KD   128
#define GLR  64
#define MM   (BS * PP)            // 16384 rows for all GEMMs

typedef __bf16 bf16x16 __attribute__((ext_vector_type(16)));
typedef float  v8f     __attribute__((ext_vector_type(8)));

union FragU { bf16x16 v; uint4 u[2]; };

typedef __attribute__((address_space(3))) void lds_void;

// Async copy of 16 bytes/lane from global to LDS (per-lane addresses).
// One wave-instruction moves 512B. Tracked by ASYNCcnt.
__device__ __forceinline__ void async_copy_b128(const __hip_bfloat16* gsrc,
                                                __hip_bfloat16* ldst) {
    unsigned lds_off = (unsigned)(unsigned long long)(lds_void*)ldst;
    asm volatile("global_load_async_to_lds_b128 %0, %1, off"
                 :
                 : "v"(lds_off), "v"(gsrc)
                 : "memory");
}

__device__ __forceinline__ void wait_async0() {
    asm volatile("s_wait_asynccnt 0" ::: "memory");
}

// Read one WMMA 16-bit operand fragment from LDS: 8 contiguous bf16 at base
// (K-half selected by lane[4] upstream) + 8 contiguous bf16 at base+16.
// Matches the CDNA5 16-bit A-matrix VGPR layout (CDNA5 ISA §7.12.2).
__device__ __forceinline__ bf16x16 lds_frag(const __hip_bfloat16* base) {
    FragU f;
    f.u[0] = *reinterpret_cast<const uint4*>(base);
    f.u[1] = *reinterpret_cast<const uint4*>(base + 16);
    return f.v;
}

__device__ __forceinline__ v8f wmma_bf16(bf16x16 a, bf16x16 b, v8f c) {
    return __builtin_amdgcn_wmma_f32_16x16x32_bf16(
        false, a, false, b, (short)0, c, false, false);
}

// ---------------------------------------------------------------------------
// f32 -> bf16 conversion (grid-stride)
// ---------------------------------------------------------------------------
__global__ void f32_to_bf16_kernel(const float* __restrict__ in,
                                   __hip_bfloat16* __restrict__ out, size_t n) {
    size_t i = (size_t)blockIdx.x * blockDim.x + threadIdx.x;
    size_t stride = (size_t)gridDim.x * blockDim.x;
    for (; i < n; i += stride) out[i] = __float2bfloat16(in[i]);
}

// ---------------------------------------------------------------------------
// WMMA GEMM: out[M,N] = A[M,K] @ W[N,K]^T (+bias) with epilogue MODE
//   MODE 0: store f32        MODE 1: store bf16      MODE 2: logsigmoid/16 f32
// Block = 256 threads = 8 waves; block tile 128x128; wave tile 32x64
// (2x4 of 16x16 accumulators). K must be a multiple of 32.
// A/B 128x32 tiles double-buffered in LDS, filled by async-to-LDS copies.
// LDS row stride padded to 40 bf16 (80B = 20 banks) -> conflict-free
// ds_load_b128 fragment reads.
// ---------------------------------------------------------------------------
template <int MODE>
__global__ __launch_bounds__(256)
void wmma_gemm(const __hip_bfloat16* __restrict__ A,
               const __hip_bfloat16* __restrict__ W,
               const float* __restrict__ bias,
               void* __restrict__ out, int M, int N, int K) {
    constexpr int LDP = 40;   // padded LDS row stride (elements)
    __shared__ __align__(16) __hip_bfloat16 sA[2][128 * LDP];
    __shared__ __align__(16) __hip_bfloat16 sB[2][128 * LDP];

    const int tid  = threadIdx.x;
    const int lane = tid & 31;
    const int wave = tid >> 5;
    const int wm   = wave & 3;        // 4 waves stacked in M
    const int wn   = wave >> 2;       // 2 waves in N
    const int mblk = blockIdx.y * 128;
    const int nblk = blockIdx.x * 128;
    const int r    = lane & 15;
    const int hs   = (lane >> 4) * 8; // K-half select within k-block

    // --- staging assignment: thread -> (row, two 8-elem chunks) ----------
    const int srow = tid >> 1;          // 0..127
    const int scol = (tid & 1) * 2;     // chunks {0,1} or {2,3} of the k-block
    const __hip_bfloat16* gA = A + (size_t)(mblk + srow) * K + scol * 8;
    int nrow = nblk + srow;
    if (nrow >= N) nrow = N - 1;        // clamp OOB columns (stores skipped)
    const __hip_bfloat16* gB = W + (size_t)nrow * K + scol * 8;
    __hip_bfloat16* lA = &sA[0][srow * LDP + scol * 8];
    __hip_bfloat16* lB = &sB[0][srow * LDP + scol * 8];
    const int bufStride = 128 * LDP;

    auto stage = [&](int buf, int kb) {
#pragma unroll
        for (int j = 0; j < 2; ++j) {
            async_copy_b128(gA + kb + j * 8, lA + buf * bufStride + j * 8);
            async_copy_b128(gB + kb + j * 8, lB + buf * bufStride + j * 8);
        }
    };

    v8f acc[2][4] = {};
    const int m0l = wm * 32;
    const int n0l = wn * 64;

    stage(0, 0);
    wait_async0();
    __syncthreads();

    for (int kb = 0; kb < K; kb += 32) {
        const int cur = (kb >> 5) & 1;
        if (kb + 32 < K) stage(cur ^ 1, kb + 32);   // overlap with compute

        const __hip_bfloat16* aBase = &sA[cur][0];
        const __hip_bfloat16* bBase = &sB[cur][0];
        bf16x16 a0 = lds_frag(aBase + (m0l + r) * LDP + hs);
        bf16x16 a1 = lds_frag(aBase + (m0l + 16 + r) * LDP + hs);
#pragma unroll
        for (int j = 0; j < 4; ++j) {
            bf16x16 bf = lds_frag(bBase + (n0l + 16 * j + r) * LDP + hs);
            acc[0][j] = wmma_bf16(a0, bf, acc[0][j]);
            acc[1][j] = wmma_bf16(a1, bf, acc[1][j]);
        }

        wait_async0();      // next buffer resident before anyone reads it
        __syncthreads();    // and everyone done reading cur before restaging
    }

    // C/D layout: lane(0-15)->col n0+lane, VGPR i -> row i (+8 for lanes 16-31)
    const int rowHalf = (lane >> 4) * 8;
#pragma unroll
    for (int it = 0; it < 2; ++it) {
        const int mrow = mblk + m0l + 16 * it + rowHalf;
#pragma unroll
        for (int j = 0; j < 4; ++j) {
            const int col = nblk + n0l + 16 * j + r;
            if (col >= N) continue;
            const float bv = bias ? bias[col] : 0.0f;
#pragma unroll
            for (int i = 0; i < 8; ++i) {
                const size_t off = (size_t)(mrow + i) * N + col;
                float val = acc[it][j][i] + bv;
                if (MODE == 1) {
                    ((__hip_bfloat16*)out)[off] = __float2bfloat16(val);
                } else if (MODE == 2) {
                    // stable log_sigmoid(x) = min(x,0) - log1p(exp(-|x|))
                    float ls = fminf(val, 0.0f) - log1pf(__expf(-fabsf(val)));
                    ((float*)out)[off] = ls * (1.0f / 16.0f);
                } else {
                    ((float*)out)[off] = val;
                }
            }
        }
    }
}

// ---------------------------------------------------------------------------
// Gated linear-attention scan. One block per (b,h); 256 threads; state
// 128x128 f32 fully register-resident (64 floats/thread). Thread owns
// column v = tid&127 for k-rows [khalf*64, khalf*64+64).
// ---------------------------------------------------------------------------
__global__ __launch_bounds__(256)
void gla_scan(const __hip_bfloat16* __restrict__ qb,
              const __hip_bfloat16* __restrict__ kb,
              const __hip_bfloat16* __restrict__ vb,
              const float* __restrict__ g,
              const float* __restrict__ state0,
              const unsigned char* __restrict__ reset,
              __hip_bfloat16* __restrict__ attn_out,
              float* __restrict__ new_state) {
    __shared__ float s_q[KD], s_k[KD], s_g[KD], s_part[KD];
    __shared__ int s_reset;

    const int tid   = threadIdx.x;
    const int b     = blockIdx.x >> 4;   // / H
    const int h     = blockIdx.x & 15;   // % H
    const int vcol  = tid & 127;
    const int khalf = tid >> 7;

    float st[64];
    const size_t sbase = ((size_t)b * HH + h) * (size_t)KD * KD;
#pragma unroll 8
    for (int kk = 0; kk < 64; ++kk)
        st[kk] = state0[sbase + (size_t)(khalf * 64 + kk) * KD + vcol];

    for (int t = 0; t < PP; ++t) {
        const size_t off = ((size_t)(b * PP + t)) * DWM + h * KD;
        if (tid < KD) {
            s_q[tid] = __bfloat162float(qb[off + tid]);
            s_k[tid] = __bfloat162float(kb[off + tid]);
            s_g[tid] = __expf(g[off + tid]);   // precompute gate factor
        }
        if (tid == 0) s_reset = reset[b * PP + t];
        __syncthreads();

        const float keep = s_reset ? 0.0f : 1.0f;
        const float vt   = __bfloat162float(vb[off + vcol]);
        float acc = 0.0f;
#pragma unroll 8
        for (int kk = 0; kk < 64; ++kk) {
            const int k = khalf * 64 + kk;
            float s = st[kk] * keep * s_g[k] + s_k[k] * vt;
            st[kk] = s;
            acc += s_q[k] * s;
        }
        if (khalf == 1) s_part[vcol] = acc;
        __syncthreads();
        if (khalf == 0)
            attn_out[off + vcol] = __float2bfloat16(acc + s_part[vcol]);
        __syncthreads();   // protect s_* before next step's loads
    }

#pragma unroll 8
    for (int kk = 0; kk < 64; ++kk)
        new_state[sbase + (size_t)(khalf * 64 + kk) * KD + vcol] = st[kk];
}

// ---------------------------------------------------------------------------
// Host-side orchestration
// ---------------------------------------------------------------------------
extern "C" void kernel_launch(void* const* d_in, const int* in_sizes, int n_in,
                              void* d_out, int out_size, void* d_ws, size_t ws_size,
                              hipStream_t stream) {
    (void)in_sizes; (void)n_in; (void)out_size; (void)ws_size;

    const float*         x      = (const float*)d_in[0];
    const unsigned char* reset  = (const unsigned char*)d_in[1];  // jax bool -> 1 byte
    const float*         state0 = (const float*)d_in[2];
    const float* Wq  = (const float*)d_in[3];  const float* bq  = (const float*)d_in[4];
    const float* Wk  = (const float*)d_in[5];  const float* bk  = (const float*)d_in[6];
    const float* Wv  = (const float*)d_in[7];  const float* bv  = (const float*)d_in[8];
    const float* Wo  = (const float*)d_in[9];  const float* bo  = (const float*)d_in[10];
    const float* Wg1 = (const float*)d_in[11];
    const float* Wg2 = (const float*)d_in[12]; const float* bg2 = (const float*)d_in[13];

    float* y_out  = (float*)d_out;                     // [32,512,2048]
    float* ns_out = y_out + (size_t)BS * PP * DD;      // [32,16,128,128]

    // bump allocator over workspace (256B aligned)
    char* ws = (char*)d_ws;
    auto alloc = [&](size_t bytes) -> char* {
        char* p = ws;
        ws += (bytes + 255) & ~(size_t)255;
        return p;
    };

    const size_t NX  = (size_t)MM * DD;        // 33.5M elems
    const size_t NW  = (size_t)DWM * DD;       // 4.2M elems
    const size_t NG1 = (size_t)GLR * DD;
    const size_t NG2 = (size_t)DWM * GLR;
    const size_t NGS = (size_t)MM * GLR;

    __hip_bfloat16* xb   = (__hip_bfloat16*)alloc(NX * 2);
    __hip_bfloat16* wqb  = (__hip_bfloat16*)alloc(NW * 2);
    __hip_bfloat16* wkb  = (__hip_bfloat16*)alloc(NW * 2);
    __hip_bfloat16* wvb  = (__hip_bfloat16*)alloc(NW * 2);
    __hip_bfloat16* wob  = (__hip_bfloat16*)alloc(NW * 2);
    __hip_bfloat16* wg1b = (__hip_bfloat16*)alloc(NG1 * 2);
    __hip_bfloat16* wg2b = (__hip_bfloat16*)alloc(NG2 * 2);
    __hip_bfloat16* qb   = (__hip_bfloat16*)alloc(NX * 2);
    __hip_bfloat16* kbuf = (__hip_bfloat16*)alloc(NX * 2);
    __hip_bfloat16* vbuf = (__hip_bfloat16*)alloc(NX * 2);
    float*          gbuf = (float*)alloc(NX * 4);
    float*          gsm  = (float*)alloc(NGS * 4);
    __hip_bfloat16* gsmb = (__hip_bfloat16*)alloc(NGS * 2);
    __hip_bfloat16* outb = (__hip_bfloat16*)alloc(NX * 2);

    // --- stage 0: f32 -> bf16 conversions -------------------------------
    f32_to_bf16_kernel<<<2048, 256, 0, stream>>>(x,   xb,   NX);
    f32_to_bf16_kernel<<<512,  256, 0, stream>>>(Wq,  wqb,  NW);
    f32_to_bf16_kernel<<<512,  256, 0, stream>>>(Wk,  wkb,  NW);
    f32_to_bf16_kernel<<<512,  256, 0, stream>>>(Wv,  wvb,  NW);
    f32_to_bf16_kernel<<<512,  256, 0, stream>>>(Wo,  wob,  NW);
    f32_to_bf16_kernel<<<64,   256, 0, stream>>>(Wg1, wg1b, NG1);
    f32_to_bf16_kernel<<<64,   256, 0, stream>>>(Wg2, wg2b, NG2);

    // --- stage 1: q/k/v projections (bf16 out, bias) --------------------
    dim3 blk(256);
    dim3 grdBig(DWM / 128, MM / 128);     // (16,128)
    wmma_gemm<1><<<grdBig, blk, 0, stream>>>(xb, wqb, bq, qb,   MM, DWM, DD);
    wmma_gemm<1><<<grdBig, blk, 0, stream>>>(xb, wkb, bk, kbuf, MM, DWM, DD);
    wmma_gemm<1><<<grdBig, blk, 0, stream>>>(xb, wvb, bv, vbuf, MM, DWM, DD);

    // --- stage 2: gate path ---------------------------------------------
    dim3 grdG1(1, MM / 128);              // N=64 (clamped inside)
    wmma_gemm<0><<<grdG1, blk, 0, stream>>>(xb, wg1b, nullptr, gsm, MM, GLR, DD);
    f32_to_bf16_kernel<<<512, 256, 0, stream>>>(gsm, gsmb, NGS);
    wmma_gemm<2><<<grdBig, blk, 0, stream>>>(gsmb, wg2b, bg2, gbuf, MM, DWM, GLR);

    // --- stage 3: gated scan, state in registers ------------------------
    gla_scan<<<dim3(BS * HH), dim3(256), 0, stream>>>(
        qb, kbuf, vbuf, gbuf, state0, reset, outb, ns_out);

    // --- stage 4: output projection (f32 straight into d_out) -----------
    wmma_gemm<0><<<grdBig, blk, 0, stream>>>(outb, wob, bo, y_out, MM, DD, DWM);
}
